// GRUModel_13975823582031
// MI455X (gfx1250) — compile-verified, matrix operand order checked by reading, not run on hardware
//
#include <hip/hip_runtime.h>

typedef __attribute__((ext_vector_type(8)))  float  v8f;
typedef __attribute__((ext_vector_type(16))) __bf16 v16bf;

#define Hdim 256
#define Vdim 8192
#define Bdim 32
#define Sdim 256
#define G3H  768   // 3*H

// ---------------------------------------------------------------------------
// Tiled transpose: W_ih [768][8192] -> W_ihT [8192][768]
// ---------------------------------------------------------------------------
__global__ __launch_bounds__(256) void transpose_wih(const float* __restrict__ in,
                                                     float* __restrict__ out) {
  __shared__ float tile[32][33];
  int bx = blockIdx.x;            // along V: 8192/32 = 256
  int by = blockIdx.y;            // along 3H: 768/32 = 24
  int tx = threadIdx.x;           // 0..31
  int ty = threadIdx.y;           // 0..7
#pragma unroll
  for (int i = 0; i < 4; ++i) {
    int r = by * 32 + ty + i * 8;               // row in W_ih (3H dim)
    int c = bx * 32 + tx;                       // col in W_ih (V dim)
    tile[ty + i * 8][tx] = in[(size_t)r * Vdim + c];
  }
  __syncthreads();
#pragma unroll
  for (int i = 0; i < 4; ++i) {
    int r = bx * 32 + ty + i * 8;               // row in W_ihT (V dim)
    int c = by * 32 + tx;                       // col in W_ihT (3H dim)
    out[(size_t)r * G3H + c] = tile[tx][ty + i * 8];
  }
}

// ---------------------------------------------------------------------------
// Coalesced embedding gather: gx[s][b][:] = W_ihT[X[b][s]][:] + b_ih
// ---------------------------------------------------------------------------
__global__ __launch_bounds__(256) void gather_gx(const int* __restrict__ X,
                                                 const float* __restrict__ W_ihT,
                                                 const float* __restrict__ b_ih,
                                                 float* __restrict__ gx) {
  int tok = blockIdx.x;           // tok = s*32 + b
  int s = tok >> 5;
  int b = tok & 31;
  int x = X[b * Sdim + s];
  const float* src = W_ihT + (size_t)x * G3H;
  float* dst = gx + (size_t)tok * G3H;
#pragma unroll
  for (int i = 0; i < 3; ++i) {
    int j = threadIdx.x + 256 * i;
    dst[j] = src[j] + b_ih[j];
  }
}

// ---------------------------------------------------------------------------
// Pack a [rows][256] f32 matrix into bf16 WMMA B-fragment (and A-fragment)
// order: dst[rt][kt][lane][e], r = rt*16 + (lane&15),
// k = kt*32 + (e&7) + ((e>>3)<<4) + 8*(lane>>4).
// One thread per (rt,kt,lane) writes 16 contiguous bf16 (32 B).
// ---------------------------------------------------------------------------
__global__ __launch_bounds__(256) void pack_frag(const float* __restrict__ src,
                                                 __bf16* __restrict__ dst) {
  int gid = blockIdx.x * 256 + threadIdx.x;
  int lane = gid & 31;
  int kt = (gid >> 5) & 7;
  int rt = gid >> 8;
  int r = rt * 16 + (lane & 15);
  int kh = (lane >> 4) << 3;
  __bf16* d = dst + (size_t)gid * 16;
  const float* s = src + (size_t)r * Hdim;
#pragma unroll
  for (int e = 0; e < 16; ++e) {
    int k = kt * 32 + (e & 7) + ((e >> 3) << 4) + kh;
    d[e] = (__bf16)s[k];
  }
}

// ---------------------------------------------------------------------------
// Persistent single-workgroup GRU scan. 1024 threads = 32 waves.
// Per step: gh = h @ W_hh.T via v_wmma_f32_16x16x32_bf16 (96 tiles, 3/wave),
// then fused gate math; h kept in LDS (f32 + bf16 shadow).
// The WhhB base pointer is laundered through an empty asm each timestep so
// LICM cannot hoist the 24 fragment loads out of the s-loop (which previously
// spilled them to scratch) — they become cheap L2-hit global loads instead.
// ---------------------------------------------------------------------------
__global__ __launch_bounds__(1024) void gru_scan(const float* __restrict__ gx,
                                                 const __bf16* __restrict__ WhhB,
                                                 const float* __restrict__ b_hh,
                                                 const float* __restrict__ h0,
                                                 float* __restrict__ Y,
                                                 float* __restrict__ h_last) {
  __shared__ float  gh[Bdim][G3H];     // 96 KB
  __shared__ float  hf[Bdim][Hdim];    // 32 KB
  __shared__ __bf16 hb[Bdim][Hdim];    // 16 KB

  int tid = threadIdx.x;
  // init state
#pragma unroll
  for (int u = 0; u < 8; ++u) {
    int idx = tid + 1024 * u;          // 0..8191
    int b = idx >> 8, j = idx & 255;
    float v = h0[idx];
    hf[b][j] = v;
    hb[b][j] = (__bf16)v;
  }
  __syncthreads();

  int wave = tid >> 5;
  int lane = tid & 31;
  int lm = lane & 15;
  int lh = (lane >> 4) << 3;

  for (int s = 0; s < Sdim; ++s) {
    // opaque copy of the weight pointer: defeats LICM/spill of B fragments
    const __bf16* wb = WhhB;
    asm volatile("" : "+v"(wb));

    // prefetch next step's gx slab (96 KB) behind the WMMA phase
    {
      int sn = (s + 1 < Sdim) ? s + 1 : s;
      const float* gnext = gx + (size_t)sn * Bdim * G3H;
      __builtin_prefetch(gnext + (size_t)tid * 24, 0, 1);
    }

    // ---- recurrent matmul: 3 output tiles per wave ----
#pragma unroll
    for (int tt = 0; tt < 3; ++tt) {
      int t = wave + 32 * tt;          // 0..95
      int mt = t / 48;                 // 0..1   (B rows)
      int nt = t % 48;                 // 0..47  (3H cols)
      v8f acc = {};
      int m = mt * 16 + lm;
      for (int kt = 0; kt < 8; ++kt) {
        v16bf a;
#pragma unroll
        for (int e = 0; e < 16; ++e) {
          int k = kt * 32 + (e & 7) + ((e >> 3) << 4) + lh;
          a[e] = hb[m][k];
        }
        v16bf bf = *reinterpret_cast<const v16bf*>(
            wb + ((size_t)((nt * 8 + kt) * 32 + lane)) * 16);
        acc = __builtin_amdgcn_wmma_f32_16x16x32_bf16(
            false, a, false, bf, (short)0, acc, false, false);
      }
#pragma unroll
      for (int i = 0; i < 8; ++i) {
        gh[mt * 16 + i + lh][nt * 16 + lm] = acc[i];
      }
    }
    __syncthreads();

    // ---- fused gates ----
    const float* gxs = gx + (size_t)s * Bdim * G3H;
#pragma unroll
    for (int u = 0; u < 8; ++u) {
      int idx = tid + 1024 * u;
      int b = idx >> 8, j = idx & 255;
      const float* g = gxs + (size_t)b * G3H;
      float ghr = gh[b][j]            + b_hh[j];
      float ghz = gh[b][Hdim + j]     + b_hh[Hdim + j];
      float ghn = gh[b][2 * Hdim + j] + b_hh[2 * Hdim + j];
      float r = 1.0f / (1.0f + __expf(-(g[j] + ghr)));
      float z = 1.0f / (1.0f + __expf(-(g[Hdim + j] + ghz)));
      float n = tanhf(g[2 * Hdim + j] + r * ghn);
      float hn = (1.0f - z) * n + z * hf[b][j];
      Y[((size_t)s * Bdim + b) * Hdim + j] = hn;
      if (s == Sdim - 1) h_last[idx] = hn;
      hf[b][j] = hn;
      hb[b][j] = (__bf16)hn;
    }
    __syncthreads();
  }
}

// ---------------------------------------------------------------------------
// Dense output GEMM: out[8192,8192] = Y @ W_dense.T + b_dense
// 256-thread block computes a 128x128 tile; 8 waves x 8 acc x 8 k-steps
// = 512 v_wmma_f32_16x16x32_bf16 per block. Fragments pre-packed.
// ---------------------------------------------------------------------------
__global__ __launch_bounds__(256) void dense_gemm(const __bf16* __restrict__ Apack,
                                                  const __bf16* __restrict__ Bpack,
                                                  const float* __restrict__ bias,
                                                  float* __restrict__ out) {
  int wave = threadIdx.x >> 5;
  int lane = threadIdx.x & 31;
  int lm = lane & 15;
  int lh = (lane >> 4) << 3;
  int mt = blockIdx.y * 8 + wave;      // 0..511 (M tiles)
  int nt0 = blockIdx.x * 8;            // 0..511 step 8 (N tiles)

  v8f acc[8] = {};
  for (int kt = 0; kt < 8; ++kt) {
    v16bf a = *reinterpret_cast<const v16bf*>(
        Apack + ((size_t)((mt * 8 + kt) * 32 + lane)) * 16);
#pragma unroll
    for (int j = 0; j < 8; ++j) {
      v16bf b = *reinterpret_cast<const v16bf*>(
          Bpack + ((size_t)(((nt0 + j) * 8 + kt) * 32 + lane)) * 16);
      acc[j] = __builtin_amdgcn_wmma_f32_16x16x32_bf16(
          false, a, false, b, (short)0, acc[j], false, false);
    }
  }
#pragma unroll
  for (int j = 0; j < 8; ++j) {
    int n = (nt0 + j) * 16 + lm;
    float bn = bias[n];
#pragma unroll
    for (int i = 0; i < 8; ++i) {
      int m = mt * 16 + i + lh;
      out[(size_t)m * Vdim + n] = acc[j][i] + bn;
    }
  }
}

// ---------------------------------------------------------------------------
extern "C" void kernel_launch(void* const* d_in, const int* in_sizes, int n_in,
                              void* d_out, int out_size, void* d_ws, size_t ws_size,
                              hipStream_t stream) {
  const int*   X       = (const int*)d_in[0];
  const float* state   = (const float*)d_in[1];
  const float* W_ih    = (const float*)d_in[2];
  const float* W_hh    = (const float*)d_in[3];
  const float* b_ih    = (const float*)d_in[4];
  const float* b_hh    = (const float*)d_in[5];
  const float* W_dense = (const float*)d_in[6];
  const float* b_dense = (const float*)d_in[7];

  float* out    = (float*)d_out;
  float* h_last = out + (size_t)Vdim * Vdim;   // second output, appended flat

  char* p = (char*)d_ws;
  float* W_ihT = (float*)p;  p += (size_t)Vdim * G3H * 4;            // 25.2 MB
  float* gx    = (float*)p;  p += (size_t)Sdim * Bdim * G3H * 4;     // 25.2 MB
  float* Ybuf  = (float*)p;  p += (size_t)Vdim * Hdim * 4;           //  8.4 MB
  __bf16* WhhB  = (__bf16*)p; p += (size_t)48  * 8 * 32 * 16 * 2;    //  0.4 MB
  __bf16* Apack = (__bf16*)p; p += (size_t)512 * 8 * 32 * 16 * 2;    //  4.2 MB
  __bf16* Bpack = (__bf16*)p; p += (size_t)512 * 8 * 32 * 16 * 2;    //  4.2 MB

  // 1. transpose W_ih for coalesced gathers
  transpose_wih<<<dim3(256, 24), dim3(32, 8), 0, stream>>>(W_ih, W_ihT);
  // 2. embedding gather + input bias
  gather_gx<<<Bdim * Sdim, 256, 0, stream>>>(X, W_ihT, b_ih, gx);
  // 3. pack W_hh.T into WMMA fragment order (48 row-tiles)
  pack_frag<<<48, 256, 0, stream>>>(W_hh, WhhB);
  // 4. pack W_dense.T into WMMA fragment order (512 row-tiles)
  pack_frag<<<512, 256, 0, stream>>>(W_dense, Bpack);
  // 5. sequential GRU scan (persistent single workgroup)
  gru_scan<<<1, 1024, 0, stream>>>(gx, WhhB, b_hh, state, Ybuf, h_last);
  // 6. pack Y into WMMA A-fragment order
  pack_frag<<<512, 256, 0, stream>>>(Ybuf, Apack);
  // 7. dense output GEMM with bias
  dense_gemm<<<dim3(64, 64), 256, 0, stream>>>(Apack, Bpack, b_dense, out);
}
